// BasicLSTM_2113123909733
// MI455X (gfx1250) — matrix operand order repaired and checked by reading, not compiled, not run on hardware
//
#include <hip/hip_runtime.h>
#include <hip/hip_bf16.h>
#include <math.h>

#define NB     16384     // batch rows
#define HID    1024      // hidden = in = out
#define KCAT   2048      // concat(h_norm, x_norm) K dimension

typedef __attribute__((ext_vector_type(16))) __bf16 v16bf;
typedef __attribute__((ext_vector_type(8)))  __bf16 v8bf;
typedef __attribute__((ext_vector_type(8)))  float  v8f;

__device__ __forceinline__ unsigned short f2bf(float f) {
    union { float f; unsigned int u; } v; v.f = f;
    unsigned int u = v.u;
    unsigned int r = u + 0x7FFFu + ((u >> 16) & 1u);   // round-to-nearest-even
    return (unsigned short)(r >> 16);
}

__device__ __forceinline__ float sigmoidf_(float x) {
    return 1.0f / (1.0f + __expf(-x));
}

// --- gfx1250 async copy: global -> LDS, 16 bytes per lane, ASYNCcnt-tracked --
__device__ __forceinline__ void async_copy16(unsigned lds_off, const void* gptr) {
    unsigned long long ga = (unsigned long long)gptr;
    asm volatile("global_load_async_to_lds_b128 %0, %1, off"
                 :: "v"(lds_off), "v"(ga) : "memory");
}
__device__ __forceinline__ void wait_async_le1() {
    asm volatile("s_wait_asynccnt 0x1" ::: "memory");
}
// Raw workgroup split-barrier WITHOUT the vmem flush __syncthreads() implies.
// The only cross-wave hand-off is async-DMA -> LDS, fenced by s_wait_asynccnt
// (producer side) before signaling; ds reads complete before each wave's
// signal by construction (their data feeds the preceding WMMAs).
__device__ __forceinline__ void barrier_ws() {
    asm volatile("s_barrier_signal -1\n\ts_barrier_wait -1" ::: "memory");
}

// ---------------------------------------------------------------------------
// Kernel 1: repack weights to bf16.
// Gate g row o: [Wh_g[o,0..1023] | Wx_g[o,0..1023]] -> Wg_cat[(g*1024+o)*2048+k]
// ---------------------------------------------------------------------------
__global__ void lstm_prep_weights(
    const float* __restrict__ Whf, const float* __restrict__ Whi,
    const float* __restrict__ Who, const float* __restrict__ Whc,
    const float* __restrict__ Wxf, const float* __restrict__ Wxi,
    const float* __restrict__ Wxo, const float* __restrict__ Wxc,
    const float* __restrict__ Why,
    unsigned short* __restrict__ Wg_cat, unsigned short* __restrict__ Why_bf)
{
    const long long idx = (long long)blockIdx.x * blockDim.x + threadIdx.x;
    const long long NG = 4LL * HID * KCAT;
    if (idx < NG) {
        int g   = (int)(idx / ((long long)HID * KCAT));
        int rem = (int)(idx % ((long long)HID * KCAT));
        int o = rem / KCAT;
        int k = rem % KCAT;
        const float* hsrc = (g == 0) ? Whf : (g == 1) ? Whi : (g == 2) ? Who : Whc;
        const float* xsrc = (g == 0) ? Wxf : (g == 1) ? Wxi : (g == 2) ? Wxo : Wxc;
        float val = (k < HID) ? hsrc[(size_t)o * HID + k]
                              : xsrc[(size_t)o * HID + (k - HID)];
        Wg_cat[idx] = f2bf(val);
    } else {
        long long j = idx - NG;
        if (j < (long long)HID * HID) Why_bf[j] = f2bf(Why[j]);
    }
}

// ---------------------------------------------------------------------------
// Kernel 2: layernorm, one wave per row; 2*NB rows.
// ---------------------------------------------------------------------------
__global__ __launch_bounds__(256) void lstm_ln(
    const float* __restrict__ x, const float* __restrict__ h_prev,
    const float* __restrict__ ln_w, const float* __restrict__ ln_b,
    const float* __restrict__ lnh_w, const float* __restrict__ lnh_b,
    unsigned short* __restrict__ A_cat)
{
    const int wave = threadIdx.x >> 5;
    const int lane = threadIdx.x & 31;
    const int job  = blockIdx.x * 8 + wave;
    const bool isH = job < NB;
    const int  row = isH ? job : job - NB;

    const float* __restrict__ src = (isH ? h_prev : x) + (size_t)row * HID;
    const float* __restrict__ w   = isH ? lnh_w : ln_w;
    const float* __restrict__ b   = isH ? lnh_b : ln_b;

    float v[32];
    float s = 0.0f;
    #pragma unroll
    for (int i = 0; i < 32; ++i) { v[i] = src[lane + 32 * i]; s += v[i]; }
    #pragma unroll
    for (int off = 16; off > 0; off >>= 1) s += __shfl_xor(s, off, 32);
    const float mu = s * (1.0f / 1024.0f);

    float q = 0.0f;
    #pragma unroll
    for (int i = 0; i < 32; ++i) { float d = v[i] - mu; q += d * d; }
    #pragma unroll
    for (int off = 16; off > 0; off >>= 1) q += __shfl_xor(q, off, 32);
    const float rstd = rsqrtf(q * (1.0f / 1024.0f) + 1e-5f);

    unsigned short* __restrict__ dst =
        A_cat + (size_t)row * KCAT + (isH ? 0 : HID);
    #pragma unroll
    for (int i = 0; i < 32; ++i) {
        int c = lane + 32 * i;
        dst[c] = f2bf((v[i] - mu) * rstd * w[c] + b[c]);
    }
}

// ---------------------------------------------------------------------------
// Kernel 3: fused 4-gate GEMM (K=2048, bf16 WMMA) + LSTM cell epilogue.
// Block = 8 waves; block tile M=256 x N=16; each wave owns 32 rows (two
// 16x16 M-subtiles) x 16 cols for all 4 gates -> 8 WMMAs per K-step.
// Weights staged in LDS via async global->LDS copies (4 KB/stage, triple
// buffered, one async instr per wave per step); each B fragment is reused
// by both M-subtiles. A fragments double-buffered in registers; raw split
// barrier keeps A prefetch in flight across stages.
// ---------------------------------------------------------------------------
__global__ __launch_bounds__(256) void lstm_gates_wmma(
    const unsigned short* __restrict__ A_cat,
    const unsigned short* __restrict__ Wg_cat,
    const float* __restrict__ bfv, const float* __restrict__ biv,
    const float* __restrict__ bov, const float* __restrict__ bcv,
    const float* __restrict__ c_prev,
    float* __restrict__ out_c, float* __restrict__ out_h,
    unsigned short* __restrict__ h_bf)
{
    __shared__ __align__(16) unsigned char smem[3][4096];

    const int t     = threadIdx.x;
    const int wave  = t >> 5;
    const int lane  = t & 31;
    const int lrow  = lane & 15;
    const int khalf = lane >> 4;
    const int rbase = blockIdx.x * 256 + wave * 32;
    const int cbase = blockIdx.y * 16;

    // async staging source for this thread (16B chunk of one weight row)
    const int srow = t >> 2, schunk = t & 3;
    const __bf16* __restrict__ wsrc = (const __bf16*)Wg_cat +
        ((size_t)((srow >> 4) * HID + cbase + (srow & 15))) * KCAT + schunk * 8;

    const __bf16* __restrict__ Arow0 =
        (const __bf16*)A_cat + (size_t)(rbase + lrow) * KCAT;
    const __bf16* __restrict__ Arow1 =
        (const __bf16*)A_cat + (size_t)(rbase + 16 + lrow) * KCAT;

    // prologue: buffer 0 <- K window 0
    async_copy16((unsigned)(unsigned long long)&smem[0][t * 16], wsrc);

    // preload A fragments for k=0
    v16bf a_cur0, a_cur1;
    {
        const v8bf alo0 = *(const v8bf*)(Arow0 + khalf * 8);
        const v8bf ahi0 = *(const v8bf*)(Arow0 + 16 + khalf * 8);
        const v8bf alo1 = *(const v8bf*)(Arow1 + khalf * 8);
        const v8bf ahi1 = *(const v8bf*)(Arow1 + 16 + khalf * 8);
        #pragma unroll
        for (int i = 0; i < 8; ++i) {
            a_cur0[i] = alo0[i]; a_cur0[i + 8] = ahi0[i];
            a_cur1[i] = alo1[i]; a_cur1[i + 8] = ahi1[i];
        }
    }

    v8f acc0[4] = {};   // M-subtile 0, gates f,i,o,c
    v8f acc1[4] = {};   // M-subtile 1
    int cur = 0;
    for (int k = 0; k < KCAT; k += 32) {
        int nxt = cur + 1; if (nxt == 3) nxt = 0;
        int kn = k + 32; if (kn >= KCAT) kn = 0;   // wrap: harmless extra stage

        // issue next weight stage (last readers of buf `nxt` passed the
        // barrier one iteration ago)
        async_copy16((unsigned)(unsigned long long)&smem[nxt][t * 16], wsrc + kn);

        // load next A fragments (register double buffer; stays in flight
        // across the raw barrier)
        v16bf a_nxt0, a_nxt1;
        {
            const v8bf alo0 = *(const v8bf*)(Arow0 + kn + khalf * 8);
            const v8bf ahi0 = *(const v8bf*)(Arow0 + kn + 16 + khalf * 8);
            const v8bf alo1 = *(const v8bf*)(Arow1 + kn + khalf * 8);
            const v8bf ahi1 = *(const v8bf*)(Arow1 + kn + 16 + khalf * 8);
            #pragma unroll
            for (int i = 0; i < 8; ++i) {
                a_nxt0[i] = alo0[i]; a_nxt0[i + 8] = ahi0[i];
                a_nxt1[i] = alo1[i]; a_nxt1[i + 8] = ahi1[i];
            }
        }

        wait_async_le1();   // current buffer complete (1 outstanding = next)
        barrier_ws();

        const unsigned char* __restrict__ bufp = smem[cur];
        #pragma unroll
        for (int g = 0; g < 4; ++g) {
            v16bf b;
            const unsigned boff = (unsigned)((g * 16 + lrow) * 64 + khalf * 16);
            const v8bf blo = *(const v8bf*)(bufp + boff);
            const v8bf bhi = *(const v8bf*)(bufp + boff + 32);
            #pragma unroll
            for (int i = 0; i < 8; ++i) { b[i] = blo[i]; b[i + 8] = bhi[i]; }
            acc0[g] = __builtin_amdgcn_wmma_f32_16x16x32_bf16(
                false, a_cur0, false, b, (short)0, acc0[g], false, false);
            acc1[g] = __builtin_amdgcn_wmma_f32_16x16x32_bf16(
                false, a_cur1, false, b, (short)0, acc1[g], false, false);
        }

        a_cur0 = a_nxt0;
        a_cur1 = a_nxt1;
        cur = nxt;
    }

    // Epilogue: full LSTM cell math on the two 16x16 tiles this wave owns.
    const int col = cbase + lrow;
    const float bF = bfv[col], bI = biv[col], bO = bov[col], bC = bcv[col];
    #pragma unroll
    for (int ms = 0; ms < 2; ++ms) {
        const v8f* acc = ms ? acc1 : acc0;
        #pragma unroll
        for (int v = 0; v < 8; ++v) {
            const int row = rbase + ms * 16 + v + (khalf << 3);
            const size_t idx = (size_t)row * HID + col;
            const float f  = sigmoidf_(acc[0][v] + bF);
            const float i  = sigmoidf_(acc[1][v] + bI);
            const float o  = sigmoidf_(acc[2][v] + bO);
            const float cp = tanhf(acc[3][v] + bC);
            const float ct = f * c_prev[idx] + i * cp;
            const float ht = o * tanhf(ct);
            out_c[idx] = ct;
            out_h[idx] = ht;
            h_bf[idx]  = f2bf(ht);
        }
    }
}

// ---------------------------------------------------------------------------
// Kernel 4: y = gelu_exact(h_t @ Why.T + by), K=1024, bf16 WMMA.
// Same structure: wave owns 32x16, async-LDS weight staging (1 KB/stage).
// ---------------------------------------------------------------------------
__global__ __launch_bounds__(256) void lstm_y_wmma(
    const unsigned short* __restrict__ h_bf,
    const unsigned short* __restrict__ Why_bf,
    const float* __restrict__ by,
    float* __restrict__ out_y)
{
    __shared__ __align__(16) unsigned char smem[3][1024];

    const int t     = threadIdx.x;
    const int wave  = t >> 5;
    const int lane  = t & 31;
    const int lrow  = lane & 15;
    const int khalf = lane >> 4;
    const int rbase = blockIdx.x * 256 + wave * 32;
    const int cbase = blockIdx.y * 16;

    const bool copier = (t < 64);
    const int srow = t >> 2, schunk = t & 3;   // valid for t<64: rows 0..15
    const __bf16* __restrict__ wsrc = (const __bf16*)Why_bf +
        (size_t)(cbase + (srow & 15)) * HID + schunk * 8;

    const __bf16* __restrict__ Arow0 =
        (const __bf16*)h_bf + (size_t)(rbase + lrow) * HID;
    const __bf16* __restrict__ Arow1 =
        (const __bf16*)h_bf + (size_t)(rbase + 16 + lrow) * HID;

    if (copier)
        async_copy16((unsigned)(unsigned long long)&smem[0][t * 16], wsrc);

    v16bf a_cur0, a_cur1;
    {
        const v8bf alo0 = *(const v8bf*)(Arow0 + khalf * 8);
        const v8bf ahi0 = *(const v8bf*)(Arow0 + 16 + khalf * 8);
        const v8bf alo1 = *(const v8bf*)(Arow1 + khalf * 8);
        const v8bf ahi1 = *(const v8bf*)(Arow1 + 16 + khalf * 8);
        #pragma unroll
        for (int i = 0; i < 8; ++i) {
            a_cur0[i] = alo0[i]; a_cur0[i + 8] = ahi0[i];
            a_cur1[i] = alo1[i]; a_cur1[i + 8] = ahi1[i];
        }
    }

    v8f acc0 = {}, acc1 = {};
    int cur = 0;
    for (int k = 0; k < HID; k += 32) {
        int nxt = cur + 1; if (nxt == 3) nxt = 0;
        int kn = k + 32; if (kn >= HID) kn = 0;

        if (copier)
            async_copy16((unsigned)(unsigned long long)&smem[nxt][t * 16], wsrc + kn);

        v16bf a_nxt0, a_nxt1;
        {
            const v8bf alo0 = *(const v8bf*)(Arow0 + kn + khalf * 8);
            const v8bf ahi0 = *(const v8bf*)(Arow0 + kn + 16 + khalf * 8);
            const v8bf alo1 = *(const v8bf*)(Arow1 + kn + khalf * 8);
            const v8bf ahi1 = *(const v8bf*)(Arow1 + kn + 16 + khalf * 8);
            #pragma unroll
            for (int i = 0; i < 8; ++i) {
                a_nxt0[i] = alo0[i]; a_nxt0[i + 8] = ahi0[i];
                a_nxt1[i] = alo1[i]; a_nxt1[i + 8] = ahi1[i];
            }
        }

        wait_async_le1();
        barrier_ws();

        {
            v16bf b;
            const unsigned boff = (unsigned)(lrow * 64 + khalf * 16);
            const v8bf blo = *(const v8bf*)(smem[cur] + boff);
            const v8bf bhi = *(const v8bf*)(smem[cur] + boff + 32);
            #pragma unroll
            for (int i = 0; i < 8; ++i) { b[i] = blo[i]; b[i + 8] = bhi[i]; }
            acc0 = __builtin_amdgcn_wmma_f32_16x16x32_bf16(
                false, a_cur0, false, b, (short)0, acc0, false, false);
            acc1 = __builtin_amdgcn_wmma_f32_16x16x32_bf16(
                false, a_cur1, false, b, (short)0, acc1, false, false);
        }

        a_cur0 = a_nxt0;
        a_cur1 = a_nxt1;
        cur = nxt;
    }

    const int col = cbase + lrow;
    const float bias = by[col];
    #pragma unroll
    for (int ms = 0; ms < 2; ++ms) {
        const v8f* acc = ms ? &acc1 : &acc0;
        #pragma unroll
        for (int v = 0; v < 8; ++v) {
            const int row = rbase + ms * 16 + v + (khalf << 3);
            const float g = (*acc)[v] + bias;
            out_y[(size_t)row * HID + col] =
                0.5f * g * (1.0f + erff(g * 0.70710678118654752f));
        }
    }
}

// ---------------------------------------------------------------------------
extern "C" void kernel_launch(void* const* d_in, const int* in_sizes, int n_in,
                              void* d_out, int out_size, void* d_ws, size_t ws_size,
                              hipStream_t stream) {
    const float* x      = (const float*)d_in[0];
    const float* c_prev = (const float*)d_in[1];
    const float* h_prev = (const float*)d_in[2];
    const float* ln_w   = (const float*)d_in[3];
    const float* ln_b   = (const float*)d_in[4];
    const float* lnh_w  = (const float*)d_in[5];
    const float* lnh_b  = (const float*)d_in[6];
    const float* Whf    = (const float*)d_in[7];
    const float* Whi    = (const float*)d_in[8];
    const float* Who    = (const float*)d_in[9];
    const float* Whc    = (const float*)d_in[10];
    const float* Wxf    = (const float*)d_in[11];
    const float* Wxi    = (const float*)d_in[12];
    const float* Wxo    = (const float*)d_in[13];
    const float* Wxc    = (const float*)d_in[14];
    const float* bfv    = (const float*)d_in[15];
    const float* biv    = (const float*)d_in[16];
    const float* bov    = (const float*)d_in[17];
    const float* bcv    = (const float*)d_in[18];
    const float* Why    = (const float*)d_in[19];
    const float* by     = (const float*)d_in[20];

    // outputs: (y, c, h) concatenated
    float* out_y = (float*)d_out;
    float* out_c = out_y + (size_t)NB * HID;
    float* out_h = out_c + (size_t)NB * HID;

    // workspace layout (bf16 buffers)
    char* ws = (char*)d_ws;
    unsigned short* A_cat  = (unsigned short*)(ws);                  // 64 MB
    unsigned short* h_bf   = (unsigned short*)(ws + (67108864));     // 32 MB
    unsigned short* Wg_cat = (unsigned short*)(ws + (100663296));    // 16 MB
    unsigned short* Why_bf = (unsigned short*)(ws + (117440512));    //  2 MB

    {
        long long total = 4LL * HID * KCAT + (long long)HID * HID;
        int blocks = (int)((total + 255) / 256);
        lstm_prep_weights<<<blocks, 256, 0, stream>>>(
            Whf, Whi, Who, Whc, Wxf, Wxi, Wxo, Wxc, Why, Wg_cat, Why_bf);
    }
    lstm_ln<<<(2 * NB) / 8, 256, 0, stream>>>(
        x, h_prev, ln_w, ln_b, lnh_w, lnh_b, A_cat);
    lstm_gates_wmma<<<dim3(NB / 256, HID / 16), 256, 0, stream>>>(
        A_cat, Wg_cat, bfv, biv, bov, bcv, c_prev, out_c, out_h, h_bf);
    lstm_y_wmma<<<dim3(NB / 256, HID / 16), 256, 0, stream>>>(
        h_bf, Why_bf, by, out_y);
}
